// GAT_41334765256978
// MI455X (gfx1250) — matrix operand order repaired and checked by reading, not compiled, not run on hardware
//
#include <hip/hip_runtime.h>
#include <stddef.h>

typedef __attribute__((ext_vector_type(16))) __bf16 v16bf;
typedef __attribute__((ext_vector_type(8)))  __bf16 v8bf;
typedef __attribute__((ext_vector_type(8)))  float  v8f;
typedef unsigned int u32;

#define F_IN 256
#define HID  128
#define NH1  8
#define C1   16
#define NC   40
#define NCP  48
#define NEG_SLOPE 0.2f
#define SM_EPS 1e-16f

// ---------- helpers ----------
__device__ __forceinline__ u32 enc_f(float f) {
    u32 b = __float_as_uint(f);
    return b ^ ((b & 0x80000000u) ? 0xFFFFFFFFu : 0x80000000u);
}
__device__ __forceinline__ float dec_f(u32 k) {
    u32 b = (k & 0x80000000u) ? (k ^ 0x80000000u) : ~k;
    return __uint_as_float(b);
}
__device__ __forceinline__ void edge_sd(const int* ei, int E, int e, int& s, int& d) {
    if (e < E) { s = ei[e]; d = ei[E + e]; }
    else       { s = e - E; d = e - E; }          // self-loops appended
}

// ---------- utility kernels ----------
__global__ void fill_u32(u32* p, u32 v, int n) {
    int i = blockIdx.x * blockDim.x + threadIdx.x;
    if (i < n) p[i] = v;
}

__global__ void cvt_bf16(const float* __restrict__ s, __bf16* __restrict__ d, int n) {
    int i = blockIdx.x * blockDim.x + threadIdx.x;
    if (i < n) d[i] = (__bf16)s[i];
}

// W1 [256,128] -> w1t [128,256] bf16 (column-major per output col = K-contiguous)
__global__ void transpose_w1(const float* __restrict__ W1, __bf16* __restrict__ w1t) {
    int i = blockIdx.x * blockDim.x + threadIdx.x;
    if (i >= HID * F_IN) return;
    int n = i / F_IN, k = i % F_IN;
    w1t[i] = (__bf16)W1[(size_t)k * HID + n];
}

// W2 [128,40] -> w2t [48,128] bf16, zero-padded cols 40..47
__global__ void transpose_w2(const float* __restrict__ W2, __bf16* __restrict__ w2t) {
    int i = blockIdx.x * blockDim.x + threadIdx.x;
    if (i >= NCP * HID) return;
    int n = i / HID, k = i % HID;
    w2t[i] = (n < NC) ? (__bf16)W2[(size_t)k * NC + n] : (__bf16)0.0f;
}

// ---------- WMMA GEMM 1: h1[N,128] = xb[N,256] @ W1  (bf16 in, f32 out) ----------
__global__ void gemm1_wmma(const __bf16* __restrict__ xb, const __bf16* __restrict__ wt,
                           float* __restrict__ h1, int N) {
    int tm = blockIdx.x, tn = blockIdx.y;
    int lane = threadIdx.x;
    int half = lane >> 4;          // 0: K 0..15, 1: K 16..31
    int l15  = lane & 15;
    int rowA = tm * 16 + l15; if (rowA >= N) rowA = N - 1;   // clamp, store guarded
    const __bf16* ap = xb + (size_t)rowA * F_IN + half * 16;
    int colB = tn * 16 + l15;
    const __bf16* bp = wt + (size_t)colB * F_IN + half * 16;

    v8f acc = {};
#pragma unroll
    for (int k0 = 0; k0 < F_IN; k0 += 32) {
        v8bf lo = *(const v8bf*)(ap + k0);        // K = k0+kk .. +7
        v8bf hi = *(const v8bf*)(ap + k0 + 8);    // K = k0+8+kk .. +7
        v16bf a = __builtin_shufflevector(lo, hi, 0,1,2,3,4,5,6,7,8,9,10,11,12,13,14,15);
        v16bf b = *(const v16bf*)(bp + k0);       // K = k0+kkb .. +15 contiguous
        acc = __builtin_amdgcn_wmma_f32_16x16x32_bf16(false, a, false, b,
                                                      (short)0, acc, false, false);
    }
    int col = tn * 16 + l15;
    int rb  = tm * 16 + half * 8;
#pragma unroll
    for (int r = 0; r < 8; r++) {
        int row = rb + r;
        if (row < N) h1[(size_t)row * HID + col] = acc[r];
    }
}

// ---------- WMMA GEMM 2: h2[N,40] = h1b[N,128] @ W2 ----------
__global__ void gemm2_wmma(const __bf16* __restrict__ xb, const __bf16* __restrict__ wt,
                           float* __restrict__ h2, int N) {
    int tm = blockIdx.x, tn = blockIdx.y;   // tn in 0..2 covers 48 padded cols
    int lane = threadIdx.x;
    int half = lane >> 4;
    int l15  = lane & 15;
    int rowA = tm * 16 + l15; if (rowA >= N) rowA = N - 1;
    const __bf16* ap = xb + (size_t)rowA * HID + half * 16;
    int colB = tn * 16 + l15;
    const __bf16* bp = wt + (size_t)colB * HID + half * 16;

    v8f acc = {};
#pragma unroll
    for (int k0 = 0; k0 < HID; k0 += 32) {
        v8bf lo = *(const v8bf*)(ap + k0);
        v8bf hi = *(const v8bf*)(ap + k0 + 8);
        v16bf a = __builtin_shufflevector(lo, hi, 0,1,2,3,4,5,6,7,8,9,10,11,12,13,14,15);
        v16bf b = *(const v16bf*)(bp + k0);
        acc = __builtin_amdgcn_wmma_f32_16x16x32_bf16(false, a, false, b,
                                                      (short)0, acc, false, false);
    }
    int col = tn * 16 + l15;
    int rb  = tm * 16 + half * 8;
#pragma unroll
    for (int r = 0; r < 8; r++) {
        int row = rb + r;
        if (row < N && col < NC) h2[(size_t)row * NC + col] = acc[r];
    }
}

// ---------- attention coefficient passes ----------
__global__ void att1(const float* __restrict__ h1, const float* __restrict__ as,
                     const float* __restrict__ ad, float* __restrict__ a_s,
                     float* __restrict__ a_d, int N) {
    int i = blockIdx.x * blockDim.x + threadIdx.x;
    if (i >= N * NH1) return;
    int n = i >> 3, h = i & 7;
    const float* hp = h1 + (size_t)n * HID + h * C1;
    float ss = 0.f, dd = 0.f;
#pragma unroll
    for (int c = 0; c < C1; c++) { float v = hp[c]; ss += v * as[h * C1 + c]; dd += v * ad[h * C1 + c]; }
    a_s[i] = ss; a_d[i] = dd;
}

__global__ void att2(const float* __restrict__ h2, const float* __restrict__ as,
                     const float* __restrict__ ad, float* __restrict__ a_s,
                     float* __restrict__ a_d, int N) {
    int n = blockIdx.x * blockDim.x + threadIdx.x;
    if (n >= N) return;
    const float* hp = h2 + (size_t)n * NC;
    float ss = 0.f, dd = 0.f;
#pragma unroll
    for (int c = 0; c < NC; c++) { float v = hp[c]; ss += v * as[c]; dd += v * ad[c]; }
    a_s[n] = ss; a_d[n] = dd;
}

// ---------- layer-1 edge passes (thread per edge*head) ----------
__global__ void edge_max1(const int* __restrict__ ei, int E, int Etot,
                          const float* __restrict__ a_s, const float* __restrict__ a_d,
                          u32* __restrict__ mx) {
    int t = blockIdx.x * blockDim.x + threadIdx.x;
    if (t >= Etot * NH1) return;
    int e = t >> 3, h = t & 7, s, d;
    edge_sd(ei, E, e, s, d);
    float ev = a_s[s * NH1 + h] + a_d[d * NH1 + h];
    ev = ev > 0.f ? ev : NEG_SLOPE * ev;
    atomicMax(&mx[d * NH1 + h], enc_f(ev));
}

__global__ void edge_sum1(const int* __restrict__ ei, int E, int Etot,
                          const float* __restrict__ a_s, const float* __restrict__ a_d,
                          const u32* __restrict__ mx, float* __restrict__ den) {
    int t = blockIdx.x * blockDim.x + threadIdx.x;
    if (t >= Etot * NH1) return;
    int e = t >> 3, h = t & 7, s, d;
    edge_sd(ei, E, e, s, d);
    float ev = a_s[s * NH1 + h] + a_d[d * NH1 + h];
    ev = ev > 0.f ? ev : NEG_SLOPE * ev;
    atomicAdd(&den[d * NH1 + h], __expf(ev - dec_f(mx[d * NH1 + h])));
}

__global__ void edge_msg1(const int* __restrict__ ei, int E, int Etot,
                          const float* __restrict__ a_s, const float* __restrict__ a_d,
                          const u32* __restrict__ mx, const float* __restrict__ den,
                          const float* __restrict__ h1, float* __restrict__ out1) {
    int t = blockIdx.x * blockDim.x + threadIdx.x;
    if (t >= Etot * NH1) return;
    int e = t >> 3, h = t & 7, s, d;
    edge_sd(ei, E, e, s, d);
    float ev = a_s[s * NH1 + h] + a_d[d * NH1 + h];
    ev = ev > 0.f ? ev : NEG_SLOPE * ev;
    float alpha = __expf(ev - dec_f(mx[d * NH1 + h])) / (den[d * NH1 + h] + SM_EPS);
    const float* hp = h1 + (size_t)s * HID + h * C1;
    float* op = out1 + (size_t)d * HID + h * C1;
#pragma unroll
    for (int c = 0; c < C1; c++) atomicAdd(op + c, hp[c] * alpha);
}

// ---------- bias + ELU + cvt to bf16 for layer 2 ----------
__global__ void fin1(const float* __restrict__ out1, const float* __restrict__ b1,
                     __bf16* __restrict__ h1b, int N) {
    int i = blockIdx.x * blockDim.x + threadIdx.x;
    if (i >= N * HID) return;
    float v = out1[i] + b1[i & (HID - 1)];
    v = v > 0.f ? v : (__expf(v) - 1.f);          // ELU
    h1b[i] = (__bf16)v;
}

// ---------- layer-2 edge passes (heads==1, thread per edge) ----------
__global__ void edge_max2(const int* __restrict__ ei, int E, int Etot,
                          const float* __restrict__ a_s, const float* __restrict__ a_d,
                          u32* __restrict__ mx) {
    int e = blockIdx.x * blockDim.x + threadIdx.x;
    if (e >= Etot) return;
    int s, d; edge_sd(ei, E, e, s, d);
    float ev = a_s[s] + a_d[d];
    ev = ev > 0.f ? ev : NEG_SLOPE * ev;
    atomicMax(&mx[d], enc_f(ev));
}

__global__ void edge_sum2(const int* __restrict__ ei, int E, int Etot,
                          const float* __restrict__ a_s, const float* __restrict__ a_d,
                          const u32* __restrict__ mx, float* __restrict__ den) {
    int e = blockIdx.x * blockDim.x + threadIdx.x;
    if (e >= Etot) return;
    int s, d; edge_sd(ei, E, e, s, d);
    float ev = a_s[s] + a_d[d];
    ev = ev > 0.f ? ev : NEG_SLOPE * ev;
    atomicAdd(&den[d], __expf(ev - dec_f(mx[d])));
}

__global__ void edge_msg2(const int* __restrict__ ei, int E, int Etot,
                          const float* __restrict__ a_s, const float* __restrict__ a_d,
                          const u32* __restrict__ mx, const float* __restrict__ den,
                          const float* __restrict__ h2, float* __restrict__ acc) {
    int e = blockIdx.x * blockDim.x + threadIdx.x;
    if (e >= Etot) return;
    int s, d; edge_sd(ei, E, e, s, d);
    float ev = a_s[s] + a_d[d];
    ev = ev > 0.f ? ev : NEG_SLOPE * ev;
    float alpha = __expf(ev - dec_f(mx[d])) / (den[d] + SM_EPS);
    const float* hp = h2 + (size_t)s * NC;
    float* op = acc + (size_t)d * NC;
#pragma unroll
    for (int c = 0; c < NC; c++) atomicAdd(op + c, hp[c] * alpha);
}

__global__ void fin2(const float* __restrict__ acc, const float* __restrict__ b2,
                     float* __restrict__ out, int N) {
    int i = blockIdx.x * blockDim.x + threadIdx.x;
    if (i >= N * NC) return;
    out[i] = acc[i] + b2[i % NC];
}

// ---------- host ----------
extern "C" void kernel_launch(void* const* d_in, const int* in_sizes, int n_in,
                              void* d_out, int out_size, void* d_ws, size_t ws_size,
                              hipStream_t stream) {
    const float* x    = (const float*)d_in[0];
    const int*   ei   = (const int*)  d_in[1];
    const float* W1   = (const float*)d_in[2];
    const float* as1  = (const float*)d_in[3];
    const float* ad1  = (const float*)d_in[4];
    const float* b1   = (const float*)d_in[5];
    const float* W2   = (const float*)d_in[6];
    const float* as2  = (const float*)d_in[7];
    const float* ad2  = (const float*)d_in[8];
    const float* b2   = (const float*)d_in[9];
    float* out = (float*)d_out;

    const int N = in_sizes[0] / F_IN;
    const int E = in_sizes[1] / 2;
    const int Etot = E + N;               // reference appends self-loops

    // workspace carve (256B aligned)
    size_t off = 0;
    char* base = (char*)d_ws;
    auto carve = [&](size_t bytes) -> char* {
        char* p = base + off;
        off = (off + bytes + 255) & ~(size_t)255;
        return p;
    };
    __bf16* xb   = (__bf16*)carve((size_t)N * F_IN * 2);
    __bf16* w1t  = (__bf16*)carve((size_t)HID * F_IN * 2);
    float*  h1   = (float*) carve((size_t)N * HID * 4);
    float*  aS1  = (float*) carve((size_t)N * NH1 * 4);
    float*  aD1  = (float*) carve((size_t)N * NH1 * 4);
    u32*    mx1  = (u32*)   carve((size_t)N * NH1 * 4);
    float*  den1 = (float*) carve((size_t)N * NH1 * 4);
    float*  out1 = (float*) carve((size_t)N * HID * 4);
    __bf16* h1b  = (__bf16*)carve((size_t)N * HID * 2);
    __bf16* w2t  = (__bf16*)carve((size_t)NCP * HID * 2);
    float*  h2   = (float*) carve((size_t)N * NC * 4);
    float*  aS2  = (float*) carve((size_t)N * 4);
    float*  aD2  = (float*) carve((size_t)N * 4);
    u32*    mx2  = (u32*)   carve((size_t)N * 4);
    float*  den2 = (float*) carve((size_t)N * 4);
    float*  acc2 = (float*) carve((size_t)N * NC * 4);

    const int B = 256;
    auto blocks = [](long long n, int b) { return (int)((n + b - 1) / b); };

    // zero accumulators (0x0 == encoded "-max" for the uint-max trick)
    fill_u32<<<blocks(N * NH1, B), B, 0, stream>>>(mx1, 0u, N * NH1);
    fill_u32<<<blocks(N * NH1, B), B, 0, stream>>>((u32*)den1, 0u, N * NH1);
    fill_u32<<<blocks((long long)N * HID, B), B, 0, stream>>>((u32*)out1, 0u, N * HID);
    fill_u32<<<blocks(N, B), B, 0, stream>>>(mx2, 0u, N);
    fill_u32<<<blocks(N, B), B, 0, stream>>>((u32*)den2, 0u, N);
    fill_u32<<<blocks((long long)N * NC, B), B, 0, stream>>>((u32*)acc2, 0u, N * NC);

    // bf16 conversions / weight transposes
    cvt_bf16<<<blocks((long long)N * F_IN, B), B, 0, stream>>>(x, xb, N * F_IN);
    transpose_w1<<<blocks(HID * F_IN, B), B, 0, stream>>>(W1, w1t);
    transpose_w2<<<blocks(NCP * HID, B), B, 0, stream>>>(W2, w2t);

    // layer 1
    dim3 g1((N + 15) / 16, HID / 16);
    gemm1_wmma<<<g1, 32, 0, stream>>>(xb, w1t, h1, N);
    att1<<<blocks(N * NH1, B), B, 0, stream>>>(h1, as1, ad1, aS1, aD1, N);
    edge_max1<<<blocks((long long)Etot * NH1, B), B, 0, stream>>>(ei, E, Etot, aS1, aD1, mx1);
    edge_sum1<<<blocks((long long)Etot * NH1, B), B, 0, stream>>>(ei, E, Etot, aS1, aD1, mx1, den1);
    edge_msg1<<<blocks((long long)Etot * NH1, B), B, 0, stream>>>(ei, E, Etot, aS1, aD1, mx1, den1, h1, out1);
    fin1<<<blocks((long long)N * HID, B), B, 0, stream>>>(out1, b1, h1b, N);

    // layer 2
    dim3 g2((N + 15) / 16, NCP / 16);
    gemm2_wmma<<<g2, 32, 0, stream>>>(h1b, w2t, h2, N);
    att2<<<blocks(N, B), B, 0, stream>>>(h2, as2, ad2, aS2, aD2, N);
    edge_max2<<<blocks(Etot, B), B, 0, stream>>>(ei, E, Etot, aS2, aD2, mx2);
    edge_sum2<<<blocks(Etot, B), B, 0, stream>>>(ei, E, Etot, aS2, aD2, mx2, den2);
    edge_msg2<<<blocks(Etot, B), B, 0, stream>>>(ei, E, Etot, aS2, aD2, mx2, den2, h2, acc2);
    fin2<<<blocks((long long)N * NC, B), B, 0, stream>>>(acc2, b2, out, N);
}
